// Block_9680856285357
// MI455X (gfx1250) — matrix operand order repaired and checked by reading, not compiled, not run on hardware
//
#include <hip/hip_runtime.h>
#include <hip/hip_bf16.h>
#include <stdint.h>

// ---------------------------------------------------------------------------
// CDNA5 (gfx1250) GPT block with top-2 MoE.
// GEMMs: v_wmma_f32_16x16x32_f16, 32x32 register tile per wave, A staged in
// LDS via the Tensor Data Mover (double buffered) when the builtin exists.
// ---------------------------------------------------------------------------

typedef __attribute__((ext_vector_type(16))) _Float16 v16h;
typedef __attribute__((ext_vector_type(8)))  _Float16 v8h;
typedef __attribute__((ext_vector_type(8)))  float    v8f;
typedef __attribute__((ext_vector_type(4)))  unsigned int v4u;
typedef __attribute__((ext_vector_type(8)))  int      v8i;
typedef __attribute__((ext_vector_type(4)))  int      v4i;

#define BB   2
#define TT   2048
#define CC   768
#define HH   12
#define DD   64
#define EE   8
#define FF   3072
#define BT   4096          // B*T
#define C3   2304          // 3*C
#define CAP  1024          // floor(2*1.0*4096/8), even, >= 4
#define ECAP (EE*CAP)

#define LDS_AROW 40        // 64B tile row + 16B TDM pad -> 16B aligned, no bank conflicts

#if defined(__HIP_DEVICE_COMPILE__) && \
    __has_builtin(__builtin_amdgcn_tensor_load_to_lds) && \
    __has_builtin(__builtin_amdgcn_s_wait_tensorcnt)
#define TDM_PATH 1
#else
#define TDM_PATH 0
#endif

static __device__ __forceinline__ v16h cat8(v8h lo, v8h hi) {
  return __builtin_shufflevector(lo, hi, 0,1,2,3,4,5,6,7,8,9,10,11,12,13,14,15);
}
static __device__ __forceinline__ v8f wmma_f16(v16h a, v16h b, v8f c) {
  return __builtin_amdgcn_wmma_f32_16x16x32_f16(false, a, false, b, (short)0, c,
                                                false, false);
}

#if TDM_PATH
// Flat LDS pointer low 32 bits == LDS byte offset (aperture maps addr[31:0]).
static __device__ __forceinline__ unsigned int lds_off(const void* p) {
  return (unsigned int)(uintptr_t)p;
}
// TDM 2D tile load: tileY rows x tileX elems (f16), row stride strideX elems,
// LDS padded +16B per 64B row (pad_interval=16 DW, pad_amount=4 DW).
static __device__ __forceinline__ void tdm_load_2d(const _Float16* gaddr,
                                                   unsigned int ldsaddr,
                                                   unsigned int tileX,
                                                   unsigned int tileY,
                                                   unsigned int strideX) {
  unsigned long long ga = (unsigned long long)(uintptr_t)gaddr;
  v4u g0;
  g0[0] = 1u;                                       // count=1, user descriptor
  g0[1] = ldsaddr;                                  // lds_addr
  g0[2] = (unsigned int)(ga & 0xffffffffu);         // global_addr[31:0]
  g0[3] = (unsigned int)((ga >> 32) & 0x01ffffffu)  // global_addr[56:32]
          | 0x80000000u;                            // type=2 ("image")
  v8i g1;
  g1[0] = (int)((1u << 16)        // data_size = 2 bytes
              | (1u << 20)        // pad_enable
              | (3u << 22)        // pad_interval: 16 DWORDs (=64B)
              | (3u << 25));      // pad_amount:   4 DWORDs (=16B)
  g1[1] = (int)((tileX & 0xffffu) << 16);                       // tensor_dim0 lo
  g1[2] = (int)(((tileX >> 16) & 0xffffu) | ((tileY & 0xffffu) << 16));
  g1[3] = (int)(((tileY >> 16) & 0xffffu) | ((tileX & 0xffffu) << 16)); // tile_dim0
  g1[4] = (int)(tileY & 0xffffu);                               // tile_dim1
  g1[5] = (int)strideX;                                         // dim0_stride lo32
  g1[6] = 0;
  g1[7] = 0;
  v4i z4 = {0, 0, 0, 0};
#if __clang_major__ >= 23
  v8i z8 = {0, 0, 0, 0, 0, 0, 0, 0};
  __builtin_amdgcn_tensor_load_to_lds(g0, g1, z4, z4, z8, 0);
#else
  __builtin_amdgcn_tensor_load_to_lds(g0, g1, z4, z4, 0);
#endif
}
#endif

// ---------------------------------------------------------------------------
// Elementwise / prep kernels
// ---------------------------------------------------------------------------

__global__ __launch_bounds__(256) void k_f32_to_f16(const float* __restrict__ in,
                                                    _Float16* __restrict__ out,
                                                    long n) {
  long i = (long)blockIdx.x * 256 + threadIdx.x;
  if (i < n) out[i] = (_Float16)in[i];
}

// in [z][R][Cc] fp32 -> out [z][Cc][R] f16   (weight -> [N,K] layout for WMMA B)
__global__ __launch_bounds__(256) void k_transpose_f16(const float* __restrict__ in,
                                                       _Float16* __restrict__ out,
                                                       int R, int Cc, long n) {
  long i = (long)blockIdx.x * 256 + threadIdx.x;
  if (i >= n) return;
  long per = (long)R * Cc;
  long z = i / per, rem = i % per;
  long c = rem / R, r = rem % R;
  out[i] = (_Float16)in[z * per + r * Cc + c];
}

// LayerNorm over C=768, one block per row, output f16
__global__ __launch_bounds__(256) void k_ln_f16(const float* __restrict__ x,
                                                const float* __restrict__ w,
                                                _Float16* __restrict__ out) {
  __shared__ float ssum[256], ssq[256];
  int row = blockIdx.x, tid = threadIdx.x;
  const float* xr = x + (long)row * CC;
  float v[3], s = 0.f, q = 0.f;
#pragma unroll
  for (int i = 0; i < 3; ++i) { v[i] = xr[tid + 256 * i]; s += v[i]; q += v[i] * v[i]; }
  ssum[tid] = s; ssq[tid] = q; __syncthreads();
  for (int st = 128; st > 0; st >>= 1) {
    if (tid < st) { ssum[tid] += ssum[tid + st]; ssq[tid] += ssq[tid + st]; }
    __syncthreads();
  }
  float mu = ssum[0] * (1.f / CC);
  float var = ssq[0] * (1.f / CC) - mu * mu;
  float rs = rsqrtf(var + 1e-5f);
#pragma unroll
  for (int i = 0; i < 3; ++i) {
    int c = tid + 256 * i;
    out[(long)row * CC + c] = (_Float16)((v[i] - mu) * rs * w[c]);
  }
}

// qkv f16 [BT,3C] -> Qh,Kh [B*H,T,D] f16 and Vt [B*H,D,T] f16
__global__ __launch_bounds__(256) void k_pack_qkv(const _Float16* __restrict__ qkv,
                                                  _Float16* __restrict__ qh,
                                                  _Float16* __restrict__ kh,
                                                  _Float16* __restrict__ vt) {
  long idx = (long)blockIdx.x * 256 + threadIdx.x;
  if (idx >= (long)BB * TT * HH * DD) return;
  int d = idx & (DD - 1);
  long r = idx >> 6;
  int h = (int)(r % HH); r /= HH;
  int t = (int)(r % TT);
  int b = (int)(r / TT);
  long row = (long)b * TT + t;
  long bh  = (long)b * HH + h;
  qh[(bh * TT + t) * DD + d] = qkv[row * C3 + h * DD + d];
  kh[(bh * TT + t) * DD + d] = qkv[row * C3 + CC + h * DD + d];
  vt[(bh * DD + d) * TT + t] = qkv[row * C3 + 2 * CC + h * DD + d];
}

// ---------------------------------------------------------------------------
// WMMA GEMM:  Out[M,N] = A[M,K](f16,row) x W[N,K](f16,row)^T
// Block = 8 waves -> 64(M) x 128(N); each wave computes 32x32 (2x2 WMMA tiles).
// A block-tile (64x32) is staged in LDS (TDM double-buffered when available),
// shared by the 4 waves of each M-row. B fragments load straight from L2.
// EPI: 0 = f32, 1 = f16, 2 = gelu->f16, 3 = f32 + residual
// ---------------------------------------------------------------------------
template <int EPI>
static __device__ __forceinline__ void store_tile(v8f acc, void* Out,
                                                  const float* Resid, long base,
                                                  int N, int row0, int col0,
                                                  int hi, int l16) {
#pragma unroll
  for (int r = 0; r < 8; ++r) {
    long o = base + (long)(row0 + r + 8 * hi) * N + col0 + l16;
    float v = acc[r];
    if (EPI == 0) ((float*)Out)[o] = v;
    else if (EPI == 1) ((_Float16*)Out)[o] = (_Float16)v;
    else if (EPI == 2) {
      float g = 0.5f * v * (1.0f + erff(v * 0.70710678118f));
      ((_Float16*)Out)[o] = (_Float16)g;
    } else {
      ((float*)Out)[o] = v + Resid[o];
    }
  }
}

template <int EPI>
__global__ __launch_bounds__(256) void k_wmma_gemm(const _Float16* __restrict__ A,
                                                   const _Float16* __restrict__ W,
                                                   void* __restrict__ Out,
                                                   const float* __restrict__ Resid,
                                                   int M, int N, int K,
                                                   long sA, long sW, long sO) {
#if TDM_PATH
  __shared__ _Float16 ldsA[2][64 * LDS_AROW];
#else
  __shared__ _Float16 ldsA[1][64 * LDS_AROW];
#endif
  int wave = threadIdx.x >> 5, lane = threadIdx.x & 31;
  int hi = lane >> 4, l16 = lane & 15;
  int wm = wave >> 2, wn = wave & 3;
  long z = blockIdx.z;
  int m0 = blockIdx.y * 64;
  int n0 = blockIdx.x * 128;
  const _Float16* Ab = A + z * sA;
  const _Float16* w0 = W + z * sW + (long)(n0 + wn * 32 + l16) * K;
  const _Float16* w1 = w0 + (long)16 * K;
  v8f acc00 = {}, acc01 = {}, acc10 = {}, acc11 = {};
  int nsteps = K >> 5;
#if TDM_PATH
  if (wave == 0)
    tdm_load_2d(Ab + (long)m0 * K, lds_off(&ldsA[0][0]), 32, 64, (unsigned)K);
#endif
  for (int i = 0; i < nsteps; ++i) {
    int k0 = i << 5;
#if TDM_PATH
    const _Float16* tile = &ldsA[i & 1][0];
    if (wave == 0) {
      if (i + 1 < nsteps) {
        tdm_load_2d(Ab + (long)m0 * K + (k0 + 32),
                    lds_off(&ldsA[(i + 1) & 1][0]), 32, 64, (unsigned)K);
        __builtin_amdgcn_s_wait_tensorcnt(1);   // tile i complete (in-order)
      } else {
        __builtin_amdgcn_s_wait_tensorcnt(0);
      }
    }
    __syncthreads();                            // tile i visible to all waves
#else
    const _Float16* tile = &ldsA[0][0];
    __syncthreads();                            // previous compute done
    {
      int r = threadIdx.x >> 2, cg = threadIdx.x & 3;
      *(v8h*)&ldsA[0][r * LDS_AROW + cg * 8] =
          *(const v8h*)(Ab + (long)(m0 + r) * K + k0 + cg * 8);
    }
    __syncthreads();
#endif
    const _Float16* ar0 = tile + (wm * 32 + l16) * LDS_AROW;
    const _Float16* ar1 = ar0 + 16 * LDS_AROW;
    v16h a0 = cat8(*(const v8h*)(ar0 + hi * 16), *(const v8h*)(ar0 + hi * 16 + 8));
    v16h a1 = cat8(*(const v8h*)(ar1 + hi * 16), *(const v8h*)(ar1 + hi * 16 + 8));
    v16h b0 = *(const v16h*)(w0 + k0 + hi * 16);
    v16h b1 = *(const v16h*)(w1 + k0 + hi * 16);
    acc00 = wmma_f16(a0, b0, acc00);
    acc01 = wmma_f16(a0, b1, acc01);
    acc10 = wmma_f16(a1, b0, acc10);
    acc11 = wmma_f16(a1, b1, acc11);
    __builtin_prefetch(w0 + k0 + 256, 0, 0);
    __builtin_prefetch(w1 + k0 + 256, 0, 0);
#if TDM_PATH
    __syncthreads();                            // compute done before overwrite
#endif
  }
  long base = z * sO;
  int mr = m0 + wm * 32, ncb = n0 + wn * 32;
  store_tile<EPI>(acc00, Out, Resid, base, N, mr,      ncb,      hi, l16);
  store_tile<EPI>(acc01, Out, Resid, base, N, mr,      ncb + 16, hi, l16);
  store_tile<EPI>(acc10, Out, Resid, base, N, mr + 16, ncb,      hi, l16);
  store_tile<EPI>(acc11, Out, Resid, base, N, mr + 16, ncb + 16, hi, l16);
}

// ---------------------------------------------------------------------------
// Flash attention: one wave per (b,h, 16-query tile). Online softmax, causal.
// ---------------------------------------------------------------------------
__global__ __launch_bounds__(256) void k_attn(const _Float16* __restrict__ Qh,
                                              const _Float16* __restrict__ Kh,
                                              const _Float16* __restrict__ Vt,
                                              _Float16* __restrict__ Oatt) {
  __shared__ _Float16 pbuf[8][16 * 34];
  int widx = threadIdx.x >> 5, lane = threadIdx.x & 31;
  int hi = lane >> 4, l16 = lane & 15;
  int wv = blockIdx.x * 8 + widx;
  int qt = wv & 127;
  int bh = wv >> 7;
  int b = bh / HH, h = bh % HH;
  const _Float16* Q = Qh + (long)bh * TT * DD;
  const _Float16* Kp = Kh + (long)bh * TT * DD;
  const _Float16* Vp = Vt + (long)bh * DD * TT;
  int q0 = qt * 16;

  const _Float16* qrow = Q + (long)(q0 + l16) * DD;
  v16h aq0 = cat8(*(const v8h*)(qrow + hi * 16),      *(const v8h*)(qrow + 8 + hi * 16));
  v16h aq1 = cat8(*(const v8h*)(qrow + 32 + hi * 16), *(const v8h*)(qrow + 40 + hi * 16));

  v8f o0 = {}, o1 = {}, o2 = {}, o3 = {};
  float mrun[8], lrun[8];
#pragma unroll
  for (int r = 0; r < 8; ++r) { mrun[r] = -3.0e38f; lrun[r] = 0.f; }

  volatile _Float16* pb = (volatile _Float16*)&pbuf[widx][0];
  int nblk = (q0 + 16 + 31) >> 5;
  for (int kb = 0; kb < nblk; ++kb) {
    int c0 = kb * 32;
    v8f s0 = {}, s1 = {};
    {
      const _Float16* kr0 = Kp + (long)(c0 + l16) * DD;
      s0 = wmma_f16(aq0, *(const v16h*)(kr0 + hi * 16), s0);
      s0 = wmma_f16(aq1, *(const v16h*)(kr0 + 32 + hi * 16), s0);
      const _Float16* kr1 = kr0 + 16 * DD;
      s1 = wmma_f16(aq0, *(const v16h*)(kr1 + hi * 16), s1);
      s1 = wmma_f16(aq1, *(const v16h*)(kr1 + 32 + hi * 16), s1);
    }
#pragma unroll
    for (int r = 0; r < 8; ++r) {
      int row = q0 + r + 8 * hi;
      int ca = c0 + l16, cb = ca + 16;
      float v0 = s0[r] * 0.125f; if (ca > row) v0 = -3.0e38f;
      float v1 = s1[r] * 0.125f; if (cb > row) v1 = -3.0e38f;
      float bm = fmaxf(v0, v1);
#pragma unroll
      for (int off = 8; off >= 1; off >>= 1) bm = fmaxf(bm, __shfl_xor(bm, off, 16));
      float mnew = fmaxf(mrun[r], bm);
      float scl = __expf(mrun[r] - mnew);
      float p0 = __expf(v0 - mnew);
      float p1 = __expf(v1 - mnew);
      float ps = p0 + p1;
#pragma unroll
      for (int off = 8; off >= 1; off >>= 1) ps += __shfl_xor(ps, off, 16);
      lrun[r] = lrun[r] * scl + ps;
      mrun[r] = mnew;
      o0[r] *= scl; o1[r] *= scl; o2[r] *= scl; o3[r] *= scl;
      pb[(r + 8 * hi) * 34 + l16] = (_Float16)p0;
      pb[(r + 8 * hi) * 34 + 16 + l16] = (_Float16)p1;
    }
    v16h ap;
#pragma unroll
    for (int i = 0; i < 8; ++i) ap[i] = pb[l16 * 34 + hi * 16 + i];
#pragma unroll
    for (int i = 0; i < 8; ++i) ap[8 + i] = pb[l16 * 34 + hi * 16 + 8 + i];
    const _Float16* vr = Vp + (long)l16 * TT + c0 + hi * 16;
    o0 = wmma_f16(ap, *(const v16h*)(vr + 0L * 16 * TT), o0);
    o1 = wmma_f16(ap, *(const v16h*)(vr + 1L * 16 * TT), o1);
    o2 = wmma_f16(ap, *(const v16h*)(vr + 2L * 16 * TT), o2);
    o3 = wmma_f16(ap, *(const v16h*)(vr + 3L * 16 * TT), o3);
  }
#pragma unroll
  for (int r = 0; r < 8; ++r) {
    float inv = 1.f / lrun[r];
    long tok = (long)b * TT + q0 + r + 8 * hi;
    long base = tok * CC + h * DD + l16;
    Oatt[base + 0]  = (_Float16)(o0[r] * inv);
    Oatt[base + 16] = (_Float16)(o1[r] * inv);
    Oatt[base + 32] = (_Float16)(o2[r] * inv);
    Oatt[base + 48] = (_Float16)(o3[r] * inv);
  }
}

// ---------------------------------------------------------------------------
// MoE gating / assignment / dispatch / combine
// ---------------------------------------------------------------------------
__global__ __launch_bounds__(256) void k_gating(const _Float16* __restrict__ xln2,
                                                const float* __restrict__ wg,
                                                int* __restrict__ tke,
                                                float* __restrict__ gp) {
  int tok = blockIdx.x * 8 + (threadIdx.x >> 5);
  int lane = threadIdx.x & 31;
  const _Float16* xr = xln2 + (long)tok * CC;
  float acc[EE] = {};
  for (int i = lane; i < CC; i += 32) {
    float xv = (float)xr[i];
#pragma unroll
    for (int e = 0; e < EE; ++e) acc[e] += xv * wg[e * CC + i];
  }
#pragma unroll
  for (int e = 0; e < EE; ++e)
#pragma unroll
    for (int off = 16; off >= 1; off >>= 1) acc[e] += __shfl_xor(acc[e], off, 32);
  if (lane == 0) {
    int i0 = 0; float v0 = acc[0];
#pragma unroll
    for (int e = 1; e < EE; ++e) if (acc[e] > v0) { v0 = acc[e]; i0 = e; }
    int i1 = 0; float v1 = -3.0e38f;
#pragma unroll
    for (int e = 0; e < EE; ++e) if (e != i0 && acc[e] > v1) { v1 = acc[e]; i1 = e; }
    float e1 = __expf(v1 - v0);
    float inv = 1.f / (1.f + e1);
    tke[tok * 2] = i0;  tke[tok * 2 + 1] = i1;
    gp[tok * 2] = inv;  gp[tok * 2 + 1] = e1 * inv;
  }
}

__global__ void k_assign(const int* __restrict__ tke, int* __restrict__ slot,
                         int* __restrict__ dtok) {
  if (blockIdx.x != 0 || threadIdx.x != 0) return;
  for (int i = 0; i < ECAP; ++i) dtok[i] = -1;
  int cnt[EE] = {};
  for (int i = 0; i < BT * 2; ++i) {
    int e = tke[i];
    int s = cnt[e]++;
    if (s < CAP) { slot[i] = s; dtok[e * CAP + s] = i >> 1; }
    else slot[i] = -1;
  }
}

__global__ __launch_bounds__(256) void k_dispatch(const int* __restrict__ dtok,
                                                  const _Float16* __restrict__ xln2,
                                                  _Float16* __restrict__ dx) {
  long row = blockIdx.x;
  int tok = dtok[row];
#pragma unroll
  for (int i = 0; i < 3; ++i) {
    int c = threadIdx.x + 256 * i;
    dx[row * CC + c] = (tok >= 0) ? xln2[(long)tok * CC + c] : (_Float16)0.f;
  }
}

__global__ __launch_bounds__(256) void k_combine(const float* __restrict__ x1,
                                                 const float* __restrict__ eout,
                                                 const int* __restrict__ tke,
                                                 const int* __restrict__ slot,
                                                 const float* __restrict__ gp,
                                                 float* __restrict__ out) {
  int t = blockIdx.x;
#pragma unroll
  for (int i = 0; i < 3; ++i) {
    int c = threadIdx.x + 256 * i;
    float acc = x1[(long)t * CC + c];
#pragma unroll
    for (int k = 0; k < 2; ++k) {
      int s = slot[t * 2 + k];
      if (s >= 0) {
        int e = tke[t * 2 + k];
        acc += gp[t * 2 + k] * eout[((long)e * CAP + s) * CC + c];
      }
    }
    out[(long)t * CC + c] = acc;
  }
}

// ---------------------------------------------------------------------------
// Host orchestration
// ---------------------------------------------------------------------------
extern "C" void kernel_launch(void* const* d_in, const int* in_sizes, int n_in,
                              void* d_out, int out_size, void* d_ws, size_t ws_size,
                              hipStream_t stream) {
  const float* x        = (const float*)d_in[0];
  const float* ln1_w    = (const float*)d_in[1];
  const float* ln2_w    = (const float*)d_in[2];
  const float* c_attn_w = (const float*)d_in[3];
  const float* c_proj_w = (const float*)d_in[4];
  const float* w_g      = (const float*)d_in[5];
  const float* c_fc     = (const float*)d_in[6];
  const float* c_proj_e = (const float*)d_in[7];

  size_t off = 0;
  auto alloc = [&](size_t bytes) -> void* {
    void* p = (char*)d_ws + off;
    off += (bytes + 255) & ~(size_t)255;
    return p;
  };
  _Float16* xln1   = (_Float16*)alloc((size_t)BT * CC * 2);
  _Float16* qkv    = (_Float16*)alloc((size_t)BT * C3 * 2);
  _Float16* qh     = (_Float16*)alloc((size_t)BB * HH * TT * DD * 2);
  _Float16* kh     = (_Float16*)alloc((size_t)BB * HH * TT * DD * 2);
  _Float16* vt     = (_Float16*)alloc((size_t)BB * HH * TT * DD * 2);
  _Float16* att    = (_Float16*)alloc((size_t)BT * CC * 2);
  float*    x1     = (float*)   alloc((size_t)BT * CC * 4);
  _Float16* xln2   = (_Float16*)alloc((size_t)BT * CC * 2);
  _Float16* wqkv16 = (_Float16*)alloc((size_t)C3 * CC * 2);
  _Float16* wproj16= (_Float16*)alloc((size_t)CC * CC * 2);
  _Float16* wfc_t  = (_Float16*)alloc((size_t)EE * FF * CC * 2);
  _Float16* wpe_t  = (_Float16*)alloc((size_t)EE * CC * FF * 2);
  int*      tke    = (int*)     alloc((size_t)BT * 2 * 4);
  float*    gp     = (float*)   alloc((size_t)BT * 2 * 4);
  int*      slot   = (int*)     alloc((size_t)BT * 2 * 4);
  int*      dtok   = (int*)     alloc((size_t)ECAP * 4);
  _Float16* dx     = (_Float16*)alloc((size_t)ECAP * CC * 2);
  _Float16* hbuf   = (_Float16*)alloc((size_t)ECAP * FF * 2);
  float*    eout   = (float*)   alloc((size_t)ECAP * CC * 4);
  (void)ws_size;

  // --- weight conversion (f32 -> f16 [N,K]) ---
  {
    long n = (long)C3 * CC;
    k_f32_to_f16<<<(n + 255) / 256, 256, 0, stream>>>(c_attn_w, wqkv16, n);
    n = (long)CC * CC;
    k_f32_to_f16<<<(n + 255) / 256, 256, 0, stream>>>(c_proj_w, wproj16, n);
    n = (long)EE * CC * FF;
    k_transpose_f16<<<(n + 255) / 256, 256, 0, stream>>>(c_fc, wfc_t, CC, FF, n);
    k_transpose_f16<<<(n + 255) / 256, 256, 0, stream>>>(c_proj_e, wpe_t, FF, CC, n);
  }

  // --- attention path ---
  k_ln_f16<<<BT, 256, 0, stream>>>(x, ln1_w, xln1);
  k_wmma_gemm<1><<<dim3(C3 / 128, BT / 64, 1), 256, 0, stream>>>(
      xln1, wqkv16, qkv, nullptr, BT, C3, CC, 0, 0, 0);
  {
    long n = (long)BB * TT * HH * DD;
    k_pack_qkv<<<(n + 255) / 256, 256, 0, stream>>>(qkv, qh, kh, vt);
  }
  k_attn<<<(BB * HH * (TT / 16)) / 8, 256, 0, stream>>>(qh, kh, vt, att);
  k_wmma_gemm<3><<<dim3(CC / 128, BT / 64, 1), 256, 0, stream>>>(
      att, wproj16, x1, x, BT, CC, CC, 0, 0, 0);

  // --- MoE path ---
  k_ln_f16<<<BT, 256, 0, stream>>>(x1, ln2_w, xln2);
  k_gating<<<BT / 8, 256, 0, stream>>>(xln2, w_g, tke, gp);
  k_assign<<<1, 1, 0, stream>>>(tke, slot, dtok);
  k_dispatch<<<ECAP, 256, 0, stream>>>(dtok, xln2, dx);
  k_wmma_gemm<2><<<dim3(FF / 128, CAP / 64, EE), 256, 0, stream>>>(
      dx, wfc_t, hbuf, nullptr, CAP, FF, CC,
      (long)CAP * CC, (long)FF * CC, (long)CAP * FF);
  k_wmma_gemm<0><<<dim3(CC / 128, CAP / 64, EE), 256, 0, stream>>>(
      hbuf, wpe_t, eout, nullptr, CAP, CC, FF,
      (long)CAP * FF, (long)CC * FF, (long)CAP * CC);
  k_combine<<<BT, 256, 0, stream>>>(x1, eout, tke, slot, gp, (float*)d_out);
}
